// DecoderLayer_69114613729788
// MI455X (gfx1250) — matrix operand order repaired and checked by reading, not compiled, not run on hardware
//
#include <hip/hip_runtime.h>
#include <math.h>
#include <stdint.h>

typedef _Float16 half_t;
typedef _Float16 v16h  __attribute__((ext_vector_type(16)));
typedef float    v8f   __attribute__((ext_vector_type(8)));
typedef uint32_t u32x4 __attribute__((ext_vector_type(4)));
typedef uint32_t u32x8 __attribute__((ext_vector_type(8)));

#define D_MODEL 512
#define D_FF    2048
#define SEQ     2048
#define NROWS   4096   // B*S = 2*2048
#define NHEADS  8
#define D_K     64

union FragU {
    v16h     v;
    uint32_t u[8];
    _Float16 h[16];
};

// ---- WMMA fragment loaders (16x16x32 f16, wave32) -------------------------
// A operand: lane = M row (lane&15); element (vgpr v, pair p):
//   K = 2v + p + 8*(half + (v>=4))
__device__ __forceinline__ v16h frag_A(const half_t* base, int lane, int stride) {
    const int m = lane & 15, hf = lane >> 4;
    const half_t* row = base + (size_t)m * stride;
    FragU f;
#pragma unroll
    for (int v = 0; v < 8; ++v) {
        int k = 2 * v + 8 * (hf + (v >> 2));       // even -> dword-aligned pair
        f.u[v] = *(const uint32_t*)(row + k);
    }
    return f.v;
}

// B operand: lane = N col (lane&15); pointer is [n][k] layout with stride.
// element (vgpr v, pair p): K = 2v + p + 8*(half + (v>=4))
__device__ __forceinline__ v16h frag_B(const half_t* base, int lane, int stride) {
    const int n = lane & 15, hf = lane >> 4;
    const half_t* col = base + (size_t)n * stride;
    FragU f;
#pragma unroll
    for (int v = 0; v < 8; ++v) {
        int k = 2 * v + 8 * (hf + (v >> 2));
        f.u[v] = *(const uint32_t*)(col + k);
    }
    return f.v;
}

__device__ __forceinline__ v8f wmma_f16(v16h a, v16h b, v8f c) {
    return __builtin_amdgcn_wmma_f32_16x16x32_f16(false, a, false, b, (short)0, c,
                                                  false, false);
}

// ---- CDNA5 async global->LDS copy (ASYNCcnt-tracked, bypasses VGPRs) ------
__device__ __forceinline__ void async_b128(half_t* ldsDst, const half_t* gSrc) {
    uint32_t lo = (uint32_t)(uintptr_t)ldsDst;   // low 32 bits = LDS byte addr
    asm volatile("global_load_async_to_lds_b128 %0, %1, off"
                 :: "v"(lo), "v"(gSrc) : "memory");
}
template <int N>
__device__ __forceinline__ void wait_async() {
#if __has_builtin(__builtin_amdgcn_s_wait_asynccnt)
    __builtin_amdgcn_s_wait_asynccnt((short)N);
#else
    asm volatile("s_wait_asynccnt %0" :: "n"(N) : "memory");
#endif
}

// ---- CDNA5 Tensor Data Mover: 2D tile DMA global->LDS ---------------------
// D# per ISA ch.8: group0 = {count, lds_addr, global_addr[56:0], type=2};
// group1 = {wg_mask=0, data_size=2B, tensor_dim0/1, tile_dim0/1, dim0_stride}.
// Groups 2/3 omitted (2D tensor). Tracked by TENSORcnt, in-order per wave.
__device__ __forceinline__ void tdm_load_2d(void* ldsDst, const void* gSrc,
                                            uint32_t tensor_d0, uint32_t tensor_d1,
                                            uint32_t tile_d0, uint32_t tile_d1,
                                            uint32_t stride0 /*elements*/) {
    uint64_t ga = (uint64_t)(uintptr_t)gSrc;
    u32x4 g0;
    g0.x = 1u;                                          // count=1, user mode
    g0.y = (uint32_t)(uintptr_t)ldsDst;                 // lds_addr (bytes)
    g0.z = (uint32_t)ga;                                // global_addr[31:0]
    g0.w = (uint32_t)((ga >> 32) & 0x01FFFFFFu) | (2u << 30);  // [56:32] | type=2
    u32x8 g1;
    g1.s0 = (1u << 16);                                 // data_size = 2 bytes
    g1.s1 = (tensor_d0 & 0xFFFFu) << 16;                // abar=0 | tdim0.lo16
    g1.s2 = (tensor_d0 >> 16) | ((tensor_d1 & 0xFFFFu) << 16);
    g1.s3 = (tensor_d1 >> 16) | (tile_d0 << 16);
    g1.s4 = tile_d1 & 0xFFFFu;                          // tile_dim1 | tile_dim2=0
    g1.s5 = stride0;                                    // dim0_stride[31:0]
    g1.s6 = 0u;                                         // stride hi | dim1_stride.lo
    g1.s7 = 0u;
    asm volatile("tensor_load_to_lds %0, %1" :: "s"(g0), "s"(g1) : "memory");
}
template <int N>
__device__ __forceinline__ void wait_tensor() {
#if __has_builtin(__builtin_amdgcn_s_wait_tensorcnt)
    __builtin_amdgcn_s_wait_tensorcnt((short)N);
#else
    asm volatile("s_wait_tensorcnt %0" :: "n"(N) : "memory");
#endif
    asm volatile("" ::: "memory");                      // keep LDS reads below
}

// ---- elementwise f32 -> f16 ----------------------------------------------
__global__ void k_cvt_f16(const float* __restrict__ in, half_t* __restrict__ out, int n) {
    int i = blockIdx.x * blockDim.x + threadIdx.x;
    if (i < n) out[i] = (half_t)in[i];
}

// ---- f32 [K][N] -> f16 transposed [N][K] (for GEMM B operands) ------------
__global__ void k_cvt_f16_t(const float* __restrict__ in, half_t* __restrict__ out,
                            int K, int N) {
    int i = blockIdx.x * blockDim.x + threadIdx.x;
    if (i < K * N) {
        int k = i / N, n = i - k * N;
        out[(size_t)n * K + k] = (half_t)in[i];
    }
}

// ---- LayerNorm (torch-style: alpha*(x-mu)/(std_ddof1+eps)+bias), f16 out --
__global__ void k_layernorm(const float* __restrict__ x, const float* __restrict__ alpha,
                            const float* __restrict__ beta, half_t* __restrict__ out) {
    __shared__ float red[128];
    const int row = blockIdx.x;
    const int tid = threadIdx.x;
    const float* xr = x + (size_t)row * D_MODEL;
    float v0 = xr[tid], v1 = xr[tid + 128], v2 = xr[tid + 256], v3 = xr[tid + 384];
    red[tid] = v0 + v1 + v2 + v3;
    __syncthreads();
    for (int o = 64; o > 0; o >>= 1) { if (tid < o) red[tid] += red[tid + o]; __syncthreads(); }
    float mu = red[0] * (1.0f / (float)D_MODEL);
    __syncthreads();
    float d0 = v0 - mu, d1 = v1 - mu, d2 = v2 - mu, d3 = v3 - mu;
    red[tid] = d0 * d0 + d1 * d1 + d2 * d2 + d3 * d3;
    __syncthreads();
    for (int o = 64; o > 0; o >>= 1) { if (tid < o) red[tid] += red[tid + o]; __syncthreads(); }
    float sd  = sqrtf(red[0] / (float)(D_MODEL - 1));   // ddof=1
    float inv = 1.0f / (sd + 1e-6f);
    half_t* orow = out + (size_t)row * D_MODEL;
    orow[tid]       = (half_t)(alpha[tid]       * d0 * inv + beta[tid]);
    orow[tid + 128] = (half_t)(alpha[tid + 128] * d1 * inv + beta[tid + 128]);
    orow[tid + 256] = (half_t)(alpha[tid + 256] * d2 * inv + beta[tid + 256]);
    orow[tid + 384] = (half_t)(alpha[tid + 384] * d3 * inv + beta[tid + 384]);
}

// ---- Tiled WMMA GEMM: C[M,N] = A[M,K] * BT[N,K]^T --------------------------
// 128 threads (4 waves); 128x128 block tile, 64x64 per wave; K-step 32.
// Double-buffered LDS tiles filled by global_load_async_to_lds_b128.
#define BM 128
#define BN 128
#define BK 32

__global__ __launch_bounds__(128)
void k_gemm(const half_t* __restrict__ A, const half_t* __restrict__ BT,
            const float* __restrict__ bias, const float* __restrict__ resid,
            float* __restrict__ outF, half_t* __restrict__ outH,
            int M, int N, int K, int relu, int transH) {
    __shared__ __align__(16) half_t sA[2][BM * BK];
    __shared__ __align__(16) half_t sB[2][BN * BK];
    const int tid  = threadIdx.x;
    const int lane = tid & 31, wv = tid >> 5;
    const int row0 = blockIdx.y * BM;
    const int col0 = blockIdx.x * BN;
    const int wr = (wv >> 1) * 64, wc = (wv & 1) * 64;

    const v8f z8 = {0.f, 0.f, 0.f, 0.f, 0.f, 0.f, 0.f, 0.f};
    v8f acc[4][4];
#pragma unroll
    for (int i = 0; i < 4; ++i)
#pragma unroll
        for (int j = 0; j < 4; ++j) acc[i][j] = z8;

    // 8 async b128 per thread per stage (A: 512 chunks, B: 512 chunks)
    auto issue = [&](int buf, int k0) {
#pragma unroll
        for (int it = 0; it < 4; ++it) {
            int c = tid + it * 128;          // 0..511
            int r = c >> 2, q = (c & 3) * 8; // 4 chunks of 8 halfs per row
            async_b128(&sA[buf][r * BK + q], A  + (size_t)(row0 + r) * K + k0 + q);
            async_b128(&sB[buf][r * BK + q], BT + (size_t)(col0 + r) * K + k0 + q);
        }
    };

    const int nk = K / BK;
    issue(0, 0);
    for (int kt = 0; kt < nk; ++kt) {
        const int buf = kt & 1;
        if (kt + 1 < nk) {
            issue(buf ^ 1, (kt + 1) * BK);
            wait_async<8>();                 // previous stage's 8 ops retired
        } else {
            wait_async<0>();
        }
        __syncthreads();
        v16h af[4], bf[4];
#pragma unroll
        for (int i = 0; i < 4; ++i) af[i] = frag_A(&sA[buf][(wr + i * 16) * BK], lane, BK);
#pragma unroll
        for (int j = 0; j < 4; ++j) bf[j] = frag_B(&sB[buf][(wc + j * 16) * BK], lane, BK);
#pragma unroll
        for (int i = 0; i < 4; ++i)
#pragma unroll
            for (int j = 0; j < 4; ++j) acc[i][j] = wmma_f16(af[i], bf[j], acc[i][j]);
        __syncthreads();                     // buffer reuse fence
    }

    const int n = lane & 15, hf = lane >> 4;
#pragma unroll
    for (int i = 0; i < 4; ++i)
#pragma unroll
        for (int j = 0; j < 4; ++j)
#pragma unroll
            for (int r = 0; r < 8; ++r) {
                int row = row0 + wr + i * 16 + r + 8 * hf;
                int col = col0 + wc + j * 16 + n;
                float v = acc[i][j][r];
                if (bias)  v += bias[col];
                if (relu)  v  = fmaxf(v, 0.0f);
                if (resid) v += resid[(size_t)row * N + col];
                if (outF)  outF[(size_t)row * N + col] = v;
                if (outH) {
                    if (transH) outH[(size_t)col * M + row] = (half_t)v;
                    else        outH[(size_t)row * N + col] = (half_t)v;
                }
            }
}

// ---- Attention: one wave per 16 query rows; scores kept in LDS ------------
// Q,K: [B*S, 512] f16 (head h at cols h*64..); Vt: [512][B*S] f16 (transposed).
// K/V operand tiles staged into LDS by the Tensor Data Mover, double-buffered.
// Replicates reference: s = QK^T/8; tril if causal; s==0 -> -inf; softmax; P@V.
__global__ __launch_bounds__(32)
void k_attn(const half_t* __restrict__ Q, const half_t* __restrict__ Kv,
            const half_t* __restrict__ Vt, half_t* __restrict__ O,
            int kvLen, int causal) {
    extern __shared__ float sW[];                     // 16 * kvLen floats (128 KB)
    __shared__ __align__(16) half_t sK[2][16 * 64];   // K tile [kv][d]
    __shared__ __align__(16) half_t sV[2][64 * 32];   // V tile [d][kv]
    __shared__ float sInv[16];
    const int lane = threadIdx.x & 31;

    const int gt = blockIdx.x;                  // 16-row tile id, 0..2047
    const int qt = gt & 127;                    // S/16 = 128 tiles per (b,h)
    const int h  = (gt >> 7) & 7;
    const int b  = gt >> 10;
    const int qrow0 = b * SEQ + qt * 16;
    const int krow0 = b * kvLen;

    const v8f z8 = {0.f, 0.f, 0.f, 0.f, 0.f, 0.f, 0.f, 0.f};
    const int n = lane & 15, hf = lane >> 4;
    const float NEG_INF = -__builtin_inff();

    const half_t* qb = Q + (size_t)qrow0 * D_MODEL + h * D_K;
    v16h aq0 = frag_A(qb, lane, D_MODEL);
    v16h aq1 = frag_A(qb + 32, lane, D_MODEL);

    // ---- S = Q K^T / 8, masked, into LDS; K tiles via TDM ----
    const half_t* kbase = Kv + (size_t)krow0 * D_MODEL + h * D_K;
    const int njt = kvLen / 16;
    tdm_load_2d(sK[0], kbase, 64, 16, 64, 16, D_MODEL);
    for (int jt = 0; jt < njt; ++jt) {
        const int buf = jt & 1;
        if (jt + 1 < njt) {
            tdm_load_2d(sK[buf ^ 1], kbase + (size_t)(jt + 1) * 16 * D_MODEL,
                        64, 16, 64, 16, D_MODEL);
            wait_tensor<1>();                   // in-order: tile jt complete
        } else {
            wait_tensor<0>();
        }
        v16h bk0 = frag_B(sK[buf], lane, 64);   // [n=kv][k=d]
        v16h bk1 = frag_B(sK[buf] + 32, lane, 64);
        v8f s = z8;
        s = wmma_f16(aq0, bk0, s);
        s = wmma_f16(aq1, bk1, s);
        int col = jt * 16 + n;
#pragma unroll
        for (int r = 0; r < 8; ++r) {
            int m = r + 8 * hf;
            int qpos = qt * 16 + m;
            float v = s[r] * 0.125f;                       // 1/sqrt(64)
            if ((causal && col > qpos) || v == 0.0f) v = NEG_INF;
            sW[(size_t)m * kvLen + col] = v;
        }
    }

    // ---- softmax per row (32 lanes cooperate per row) ----
    for (int m = 0; m < 16; ++m) {
        float* rowp = sW + (size_t)m * kvLen;
        float mx = NEG_INF;
        for (int c = lane; c < kvLen; c += 32) mx = fmaxf(mx, rowp[c]);
        for (int o = 16; o > 0; o >>= 1) mx = fmaxf(mx, __shfl_xor(mx, o, 32));
        float sm = 0.0f;
        for (int c = lane; c < kvLen; c += 32) {
            float e = __expf(rowp[c] - mx);
            rowp[c] = e;
            sm += e;
        }
        for (int o = 16; o > 0; o >>= 1) sm += __shfl_xor(sm, o, 32);
        if (lane == 0) sInv[m] = 1.0f / sm;
    }

    // ---- O = P @ V; V tiles (transposed layout) via TDM ----
    v8f co[4] = {z8, z8, z8, z8};
    const int mA = lane & 15;
    const half_t* vbase = Vt + (size_t)(h * D_K) * NROWS + krow0;
    const int nkt = kvLen / 32;
    tdm_load_2d(sV[0], vbase, 32, 64, 32, 64, NROWS);
    for (int kt = 0; kt < nkt; ++kt) {
        const int buf = kt & 1;
        if (kt + 1 < nkt) {
            tdm_load_2d(sV[buf ^ 1], vbase + (kt + 1) * 32, 32, 64, 32, 64, NROWS);
            wait_tensor<1>();
        } else {
            wait_tensor<0>();
        }
        FragU pf;                                   // P as A operand (f32->f16)
#pragma unroll
        for (int v = 0; v < 8; ++v) {
            int k = 2 * v + 8 * (hf + (v >> 2));
            pf.h[2 * v]     = (_Float16)sW[(size_t)mA * kvLen + kt * 32 + k];
            pf.h[2 * v + 1] = (_Float16)sW[(size_t)mA * kvLen + kt * 32 + k + 1];
        }
#pragma unroll
        for (int g = 0; g < 4; ++g) {
            v16h vf = frag_B(sV[buf] + (size_t)(g * 16) * 32, lane, 32); // [n=d][k=kv]
            co[g] = wmma_f16(pf.v, vf, co[g]);
        }
    }

#pragma unroll
    for (int g = 0; g < 4; ++g)
#pragma unroll
        for (int r = 0; r < 8; ++r) {
            int m = r + 8 * hf;
            float val = co[g][r] * sInv[m];
            O[(size_t)(qrow0 + m) * D_MODEL + h * D_K + g * 16 + n] = (half_t)val;
        }
}

// ---------------------------------------------------------------------------
extern "C" void kernel_launch(void* const* d_in, const int* in_sizes, int n_in,
                              void* d_out, int out_size, void* d_ws, size_t ws_size,
                              hipStream_t stream) {
    const float* x      = (const float*)d_in[0];
    const float* memory = (const float*)d_in[1];
    const float* ln1_a  = (const float*)d_in[2];
    const float* ln1_b  = (const float*)d_in[3];
    const float* ln2_a  = (const float*)d_in[4];
    const float* ln2_b  = (const float*)d_in[5];
    const float* ln3_a  = (const float*)d_in[6];
    const float* ln3_b  = (const float*)d_in[7];
    const float* a1_wq  = (const float*)d_in[8];
    const float* a1_wk  = (const float*)d_in[9];
    const float* a1_wv  = (const float*)d_in[10];
    const float* a1_wo  = (const float*)d_in[11];
    const float* a1_bo  = (const float*)d_in[12];
    const float* a2_wq  = (const float*)d_in[13];
    const float* a2_wk  = (const float*)d_in[14];
    const float* a2_wv  = (const float*)d_in[15];
    const float* a2_wo  = (const float*)d_in[16];
    const float* a2_bo  = (const float*)d_in[17];
    const float* ff_w1  = (const float*)d_in[18];
    const float* ff_b1  = (const float*)d_in[19];
    const float* ff_w2  = (const float*)d_in[20];
    const float* ff_b2  = (const float*)d_in[21];
    float* out = (float*)d_out;

    char* base = (char*)d_ws;
    size_t off = 0;
    auto alloc = [&](size_t bytes) {
        char* p = base + off;
        off += (bytes + 255) & ~(size_t)255;
        return p;
    };
    half_t* h_x2  = (half_t*)alloc((size_t)NROWS * D_MODEL * 2);
    half_t* h_mem = (half_t*)alloc((size_t)NROWS * D_MODEL * 2);
    half_t* h_q   = (half_t*)alloc((size_t)NROWS * D_MODEL * 2);
    half_t* h_k   = (half_t*)alloc((size_t)NROWS * D_MODEL * 2);
    half_t* h_vT  = (half_t*)alloc((size_t)NROWS * D_MODEL * 2);  // [512][4096]
    half_t* h_ao  = (half_t*)alloc((size_t)NROWS * D_MODEL * 2);
    half_t* h_ff  = (half_t*)alloc((size_t)NROWS * D_FF * 2);
    float*  f_xa  = (float*)alloc((size_t)NROWS * D_MODEL * 4);
    float*  f_xb  = (float*)alloc((size_t)NROWS * D_MODEL * 4);
    // transposed f16 weights: WT[n][k]
    half_t* h_wq1 = (half_t*)alloc((size_t)D_MODEL * D_MODEL * 2);
    half_t* h_wk1 = (half_t*)alloc((size_t)D_MODEL * D_MODEL * 2);
    half_t* h_wv1 = (half_t*)alloc((size_t)D_MODEL * D_MODEL * 2);
    half_t* h_wo1 = (half_t*)alloc((size_t)D_MODEL * D_MODEL * 2);
    half_t* h_wq2 = (half_t*)alloc((size_t)D_MODEL * D_MODEL * 2);
    half_t* h_wk2 = (half_t*)alloc((size_t)D_MODEL * D_MODEL * 2);
    half_t* h_wv2 = (half_t*)alloc((size_t)D_MODEL * D_MODEL * 2);
    half_t* h_wo2 = (half_t*)alloc((size_t)D_MODEL * D_MODEL * 2);
    half_t* h_w1  = (half_t*)alloc((size_t)D_MODEL * D_FF * 2);   // [2048][512]
    half_t* h_w2  = (half_t*)alloc((size_t)D_FF * D_MODEL * 2);   // [512][2048]
    (void)ws_size; (void)n_in; (void)in_sizes; (void)out_size;

    auto cvtT = [&](const float* src, half_t* dst, int K, int N) {
        k_cvt_f16_t<<<(K * N + 255) / 256, 256, 0, stream>>>(src, dst, K, N);
    };
    cvtT(a1_wq, h_wq1, D_MODEL, D_MODEL);  cvtT(a1_wk, h_wk1, D_MODEL, D_MODEL);
    cvtT(a1_wv, h_wv1, D_MODEL, D_MODEL);  cvtT(a1_wo, h_wo1, D_MODEL, D_MODEL);
    cvtT(a2_wq, h_wq2, D_MODEL, D_MODEL);  cvtT(a2_wk, h_wk2, D_MODEL, D_MODEL);
    cvtT(a2_wv, h_wv2, D_MODEL, D_MODEL);  cvtT(a2_wo, h_wo2, D_MODEL, D_MODEL);
    cvtT(ff_w1, h_w1, D_MODEL, D_FF);
    cvtT(ff_w2, h_w2, D_FF, D_MODEL);
    k_cvt_f16<<<(NROWS * D_MODEL + 255) / 256, 256, 0, stream>>>(memory, h_mem,
                                                                 NROWS * D_MODEL);

    auto gemm = [&](const half_t* A, const half_t* BT, const float* bias,
                    const float* resid, float* oF, half_t* oH,
                    int M, int N, int K, int relu, int transH) {
        dim3 grid(N / BN, M / BM);
        k_gemm<<<grid, dim3(128), 0, stream>>>(A, BT, bias, resid, oF, oH,
                                               M, N, K, relu, transH);
    };

    const size_t attnShm = 16u * SEQ * 4u;   // 128 KB score rows per workgroup

    // ---- stage 1: self-attention (causal) ----
    k_layernorm<<<NROWS, 128, 0, stream>>>(x, ln1_a, ln1_b, h_x2);
    gemm(h_x2, h_wq1, nullptr, nullptr, nullptr, h_q,  NROWS, D_MODEL, D_MODEL, 0, 0);
    gemm(h_x2, h_wk1, nullptr, nullptr, nullptr, h_k,  NROWS, D_MODEL, D_MODEL, 0, 0);
    gemm(h_x2, h_wv1, nullptr, nullptr, nullptr, h_vT, NROWS, D_MODEL, D_MODEL, 0, 1);
    k_attn<<<2048, 32, attnShm, stream>>>(h_q, h_k, h_vT, h_ao, SEQ, 1);
    gemm(h_ao, h_wo1, a1_bo, x, f_xa, nullptr, NROWS, D_MODEL, D_MODEL, 0, 0);

    // ---- stage 2: cross-attention ----
    k_layernorm<<<NROWS, 128, 0, stream>>>(f_xa, ln2_a, ln2_b, h_x2);
    gemm(h_x2,  h_wq2, nullptr, nullptr, nullptr, h_q,  NROWS, D_MODEL, D_MODEL, 0, 0);
    gemm(h_mem, h_wk2, nullptr, nullptr, nullptr, h_k,  NROWS, D_MODEL, D_MODEL, 0, 0);
    gemm(h_mem, h_wv2, nullptr, nullptr, nullptr, h_vT, NROWS, D_MODEL, D_MODEL, 0, 1);
    k_attn<<<2048, 32, attnShm, stream>>>(h_q, h_k, h_vT, h_ao, SEQ, 0);
    gemm(h_ao, h_wo2, a2_bo, f_xa, f_xb, nullptr, NROWS, D_MODEL, D_MODEL, 0, 0);

    // ---- stage 3: feed-forward ----
    k_layernorm<<<NROWS, 128, 0, stream>>>(f_xb, ln3_a, ln3_b, h_x2);
    gemm(h_x2, h_w1, ff_b1, nullptr, nullptr, h_ff, NROWS, D_FF, D_MODEL, 1, 0);
    gemm(h_ff, h_w2, ff_b2, f_xb, out, nullptr, NROWS, D_MODEL, D_FF, 0, 0);
}
